// DecoderRNN_37306085933466
// MI455X (gfx1250) — matrix operand order repaired and checked by reading, not compiled, not run on hardware
//
#include <hip/hip_runtime.h>
#include <math.h>

// ---------------- problem constants ----------------
#define B_  32
#define T_  128
#define E_  512
#define H_  1024
#define V_  32000
#define BOS_ 1

// ---------------- types ----------------
typedef __bf16 bf16;
typedef __attribute__((ext_vector_type(8)))  bf16  bf16x8;
typedef __attribute__((ext_vector_type(16))) bf16  bf16x16;
typedef __attribute__((ext_vector_type(8)))  float f32x8;

// ---------------- WMMA helpers (CDNA5 16x16x32 bf16, wave32) ----------------
__device__ __forceinline__ bf16x16 cat16(bf16x8 lo, bf16x8 hi) {
  return __builtin_shufflevector(lo, hi, 0,1,2,3,4,5,6,7,8,9,10,11,12,13,14,15);
}

// A fragment: 16x32 bf16 tile, row-major, one row per lane (row = lane&15).
//   elems [0..7]  = A[row][kk + 8*half       .. +7]
//   elems [8..15] = A[row][kk + 16 + 8*half  .. +7]
__device__ __forceinline__ bf16x16 load_fragA(const bf16* Arow, int kk, int half) {
  bf16x8 lo = *(const bf16x8*)(Arow + kk + half * 8);
  bf16x8 hi = *(const bf16x8*)(Arow + kk + 16 + half * 8);
  return cat16(lo, hi);
}

// B fragment: 32x16 (K x N); source is W row-major [N, K] (B[k][n] = W[n][k]).
// Lane holds column n = lane&15; elems [0..7] K=kk+8*half..+7, elems [8..15] K=kk+16+8*half..+7.
__device__ __forceinline__ bf16x16 load_fragB(const bf16* Wrow, int kk, int half) {
  bf16x8 lo = *(const bf16x8*)(Wrow + kk + half * 8);
  bf16x8 hi = *(const bf16x8*)(Wrow + kk + 16 + half * 8);
  return cat16(lo, hi);
}

__device__ __forceinline__ f32x8 wmma_bf16(bf16x16 a, bf16x16 b, f32x8 c) {
  return __builtin_amdgcn_wmma_f32_16x16x32_bf16(false, a, false, b, (short)0, c, false, false);
}

// ---------------- small utility kernels ----------------
__global__ void k_f32_to_bf16(const float* __restrict__ src, bf16* __restrict__ dst, size_t n) {
  size_t i = (size_t)blockIdx.x * blockDim.x + threadIdx.x;
  size_t stride = (size_t)gridDim.x * blockDim.x;
  for (; i < n; i += stride) dst[i] = (bf16)src[i];
}

__global__ void k_copy_f32(const float* __restrict__ src, float* __restrict__ dst, size_t n) {
  size_t i = (size_t)blockIdx.x * blockDim.x + threadIdx.x;
  size_t stride = (size_t)gridDim.x * blockDim.x;
  for (; i < n; i += stride) dst[i] = src[i];
}

__global__ void k_init_h(const float* __restrict__ eh, float* __restrict__ hf, bf16* __restrict__ hb) {
  size_t n = (size_t)B_ * H_;
  size_t i = (size_t)blockIdx.x * blockDim.x + threadIdx.x;
  size_t stride = (size_t)gridDim.x * blockDim.x;
  for (; i < n; i += stride) { float v = eh[i]; hf[i] = v; hb[i] = (bf16)v; }
}

// X[t*B + b][e] = bf16(relu(embedding[tok])) ; tok = (t==0) ? BOS : target[b][t-1]
__global__ void k_embed(const int* __restrict__ target, const float* __restrict__ emb,
                        bf16* __restrict__ X) {
  int tb = blockIdx.x;            // tb = t*B + b
  int t = tb / B_, b = tb % B_;
  int tok = (t == 0) ? BOS_ : target[b * T_ + (t - 1)];
  const float* e = emb + (size_t)tok * E_;
  bf16* xo = X + (size_t)tb * E_;
  for (int i = threadIdx.x; i < E_; i += blockDim.x)
    xo[i] = (bf16)fmaxf(e[i], 0.0f);
}

// ---------------- Gi = X @ W_ih^T + b_ih  (4096 x 3072 x 512) ----------------
// One wave -> 32x64 output tile (2 mtiles x 4 ntiles, 8 WMMA accumulators),
// with software-pipelined (double-buffered) fragment loads.
__global__ __launch_bounds__(128)
void k_gi_gemm(const bf16* __restrict__ X, const bf16* __restrict__ Wih,
               const float* __restrict__ bih, float* __restrict__ Gi) {
  int wave = threadIdx.x >> 5, lane = threadIdx.x & 31;
  int wcol = blockIdx.x * 4 + wave;   // 0..47, each = 4 ntiles (64 cols)
  int mt0  = blockIdx.y * 2;          // 2 mtiles (32 rows)
  int half = lane >> 4, l = lane & 15;

  const bf16* Arow0 = X + (size_t)(mt0 * 16 + l) * E_;
  const bf16* Arow1 = Arow0 + (size_t)16 * E_;
  int nc0 = wcol * 64 + l;
  const bf16* B0 = Wih + (size_t)(nc0 +  0) * E_;
  const bf16* B1 = Wih + (size_t)(nc0 + 16) * E_;
  const bf16* B2 = Wih + (size_t)(nc0 + 32) * E_;
  const bf16* B3 = Wih + (size_t)(nc0 + 48) * E_;

  f32x8 acc[2][4] = {};
  bf16x16 a0 = load_fragA(Arow0, 0, half);
  bf16x16 a1 = load_fragA(Arow1, 0, half);
  bf16x16 b0 = load_fragB(B0, 0, half);
  bf16x16 b1 = load_fragB(B1, 0, half);
  bf16x16 b2 = load_fragB(B2, 0, half);
  bf16x16 b3 = load_fragB(B3, 0, half);
  for (int kk = 0; kk < E_; kk += 32) {
    int nk = (kk + 32) & (E_ - 1);        // wrap: final prefetch is redundant, harmless
    bf16x16 na0 = load_fragA(Arow0, nk, half);
    bf16x16 na1 = load_fragA(Arow1, nk, half);
    bf16x16 nb0 = load_fragB(B0, nk, half);
    bf16x16 nb1 = load_fragB(B1, nk, half);
    bf16x16 nb2 = load_fragB(B2, nk, half);
    bf16x16 nb3 = load_fragB(B3, nk, half);
    acc[0][0] = wmma_bf16(a0, b0, acc[0][0]);
    acc[1][0] = wmma_bf16(a1, b0, acc[1][0]);
    acc[0][1] = wmma_bf16(a0, b1, acc[0][1]);
    acc[1][1] = wmma_bf16(a1, b1, acc[1][1]);
    acc[0][2] = wmma_bf16(a0, b2, acc[0][2]);
    acc[1][2] = wmma_bf16(a1, b2, acc[1][2]);
    acc[0][3] = wmma_bf16(a0, b3, acc[0][3]);
    acc[1][3] = wmma_bf16(a1, b3, acc[1][3]);
    a0 = na0; a1 = na1; b0 = nb0; b1 = nb1; b2 = nb2; b3 = nb3;
  }
  #pragma unroll
  for (int j = 0; j < 4; ++j) {
    int ncol = nc0 + 16 * j;
    float bias = bih[ncol];
    #pragma unroll
    for (int mi = 0; mi < 2; ++mi) {
      #pragma unroll
      for (int i = 0; i < 8; ++i) {
        int m = (mt0 + mi) * 16 + i + 8 * half;
        Gi[(size_t)m * (3 * H_) + ncol] = acc[mi][j][i] + bias;
      }
    }
  }
}

// ---------------- one GRU step ----------------
// One wave -> one 16-col ntile, both batch mtiles, all 3 gates (6 accumulators),
// software-pipelined fragment loads.
__global__ __launch_bounds__(256)
void k_gru_step(const bf16* __restrict__ hbf_t, const float* __restrict__ hf_t,
                const bf16* __restrict__ Whh, const float* __restrict__ bhh,
                const float* __restrict__ Gi_t,
                float* __restrict__ hf_next, bf16* __restrict__ hbf_next) {
  int wave = threadIdx.x >> 5, lane = threadIdx.x & 31;
  int nt = blockIdx.x * 8 + wave;    // 0..63
  int half = lane >> 4, l = lane & 15;

  const bf16* Arow0 = hbf_t + (size_t)l * H_;          // rows 0..15
  const bf16* Arow1 = Arow0 + (size_t)16 * H_;         // rows 16..31
  int ncol = nt * 16 + l;
  const bf16* Br = Whh + (size_t)(0 * H_ + ncol) * H_;
  const bf16* Bz = Whh + (size_t)(1 * H_ + ncol) * H_;
  const bf16* Bn = Whh + (size_t)(2 * H_ + ncol) * H_;

  f32x8 aR[2] = {}, aZ[2] = {}, aN[2] = {};
  bf16x16 a0 = load_fragA(Arow0, 0, half);
  bf16x16 a1 = load_fragA(Arow1, 0, half);
  bf16x16 br = load_fragB(Br, 0, half);
  bf16x16 bz = load_fragB(Bz, 0, half);
  bf16x16 bn = load_fragB(Bn, 0, half);
  for (int kk = 0; kk < H_; kk += 32) {
    int nk = (kk + 32) & (H_ - 1);
    bf16x16 na0 = load_fragA(Arow0, nk, half);
    bf16x16 na1 = load_fragA(Arow1, nk, half);
    bf16x16 nbr = load_fragB(Br, nk, half);
    bf16x16 nbz = load_fragB(Bz, nk, half);
    bf16x16 nbn = load_fragB(Bn, nk, half);
    aR[0] = wmma_bf16(a0, br, aR[0]);
    aR[1] = wmma_bf16(a1, br, aR[1]);
    aZ[0] = wmma_bf16(a0, bz, aZ[0]);
    aZ[1] = wmma_bf16(a1, bz, aZ[1]);
    aN[0] = wmma_bf16(a0, bn, aN[0]);
    aN[1] = wmma_bf16(a1, bn, aN[1]);
    a0 = na0; a1 = na1; br = nbr; bz = nbz; bn = nbn;
  }

  float br_b = bhh[ncol], bz_b = bhh[H_ + ncol], bn_b = bhh[2 * H_ + ncol];
  #pragma unroll
  for (int mi = 0; mi < 2; ++mi) {
    #pragma unroll
    for (int i = 0; i < 8; ++i) {
      int row = mi * 16 + i + 8 * half;           // batch index b
      size_t g = (size_t)row * (3 * H_);
      float r  = 1.0f / (1.0f + __expf(-(Gi_t[g + ncol] + aR[mi][i] + br_b)));
      float z  = 1.0f / (1.0f + __expf(-(Gi_t[g + H_ + ncol] + aZ[mi][i] + bz_b)));
      float hn = aN[mi][i] + bn_b;
      float n  = tanhf(Gi_t[g + 2 * H_ + ncol] + r * hn);
      float hp = hf_t[(size_t)row * H_ + ncol];
      float hnew = (1.0f - z) * n + z * hp;
      hf_next[(size_t)row * H_ + ncol]  = hnew;
      hbf_next[(size_t)row * H_ + ncol] = (bf16)hnew;
    }
  }
}

// ---------------- logits = H @ W_out^T + b_out  (4096 x 32000 x 1024) ----------------
// One wave -> 32x64 output tile, software-pipelined fragment loads.
__global__ __launch_bounds__(128)
void k_out_gemm(const bf16* __restrict__ Hbf, const bf16* __restrict__ Wout,
                const float* __restrict__ bout, float* __restrict__ out) {
  int wave = threadIdx.x >> 5, lane = threadIdx.x & 31;
  int wcol = blockIdx.x * 4 + wave;   // 0..499 (each = 64 cols)
  int mt0  = blockIdx.y * 2;          // 32 rows
  int half = lane >> 4, l = lane & 15;

  const bf16* Arow0 = Hbf + (size_t)(mt0 * 16 + l) * H_;
  const bf16* Arow1 = Arow0 + (size_t)16 * H_;
  int nc0 = wcol * 64 + l;
  const bf16* B0 = Wout + (size_t)(nc0 +  0) * H_;
  const bf16* B1 = Wout + (size_t)(nc0 + 16) * H_;
  const bf16* B2 = Wout + (size_t)(nc0 + 32) * H_;
  const bf16* B3 = Wout + (size_t)(nc0 + 48) * H_;

  f32x8 acc[2][4] = {};
  bf16x16 a0 = load_fragA(Arow0, 0, half);
  bf16x16 a1 = load_fragA(Arow1, 0, half);
  bf16x16 b0 = load_fragB(B0, 0, half);
  bf16x16 b1 = load_fragB(B1, 0, half);
  bf16x16 b2 = load_fragB(B2, 0, half);
  bf16x16 b3 = load_fragB(B3, 0, half);
  for (int kk = 0; kk < H_; kk += 32) {
    int nk = (kk + 32) & (H_ - 1);
    bf16x16 na0 = load_fragA(Arow0, nk, half);
    bf16x16 na1 = load_fragA(Arow1, nk, half);
    bf16x16 nb0 = load_fragB(B0, nk, half);
    bf16x16 nb1 = load_fragB(B1, nk, half);
    bf16x16 nb2 = load_fragB(B2, nk, half);
    bf16x16 nb3 = load_fragB(B3, nk, half);
    acc[0][0] = wmma_bf16(a0, b0, acc[0][0]);
    acc[1][0] = wmma_bf16(a1, b0, acc[1][0]);
    acc[0][1] = wmma_bf16(a0, b1, acc[0][1]);
    acc[1][1] = wmma_bf16(a1, b1, acc[1][1]);
    acc[0][2] = wmma_bf16(a0, b2, acc[0][2]);
    acc[1][2] = wmma_bf16(a1, b2, acc[1][2]);
    acc[0][3] = wmma_bf16(a0, b3, acc[0][3]);
    acc[1][3] = wmma_bf16(a1, b3, acc[1][3]);
    a0 = na0; a1 = na1; b0 = nb0; b1 = nb1; b2 = nb2; b3 = nb3;
  }
  #pragma unroll
  for (int j = 0; j < 4; ++j) {
    int ncol = nc0 + 16 * j;
    float bias = bout[ncol];
    #pragma unroll
    for (int mi = 0; mi < 2; ++mi) {
      #pragma unroll
      for (int i = 0; i < 8; ++i) {
        int m = (mt0 + mi) * 16 + i + 8 * half;   // t*B + b
        int t = m / B_, b = m % B_;
        out[((size_t)b * T_ + t) * V_ + ncol] = acc[mi][j][i] + bias;
      }
    }
  }
}

// ---------------- in-place log-softmax over V per (b,t) row ----------------
__global__ __launch_bounds__(256)
void k_logsoftmax(float* __restrict__ out) {
  float* x = out + (size_t)blockIdx.x * V_;
  __shared__ float red[256];
  float m = -INFINITY;
  for (int i = threadIdx.x; i < V_; i += 256) m = fmaxf(m, x[i]);
  red[threadIdx.x] = m; __syncthreads();
  for (int s = 128; s > 0; s >>= 1) {
    if (threadIdx.x < s) red[threadIdx.x] = fmaxf(red[threadIdx.x], red[threadIdx.x + s]);
    __syncthreads();
  }
  m = red[0]; __syncthreads();
  float sum = 0.0f;
  for (int i = threadIdx.x; i < V_; i += 256) sum += __expf(x[i] - m);
  red[threadIdx.x] = sum; __syncthreads();
  for (int s = 128; s > 0; s >>= 1) {
    if (threadIdx.x < s) red[threadIdx.x] += red[threadIdx.x + s];
    __syncthreads();
  }
  float lse = m + logf(red[0]);
  for (int i = threadIdx.x; i < V_; i += 256) x[i] -= lse;
}

// ---------------- host launcher ----------------
extern "C" void kernel_launch(void* const* d_in, const int* in_sizes, int n_in,
                              void* d_out, int out_size, void* d_ws, size_t ws_size,
                              hipStream_t stream) {
  const float* enc_hidden = (const float*)d_in[1];   // [1,B,H]
  const int*   target     = (const int*)  d_in[2];   // [B,T]
  const float* embedding  = (const float*)d_in[3];   // [V,E]
  const float* W_ih       = (const float*)d_in[4];   // [3H,E]
  const float* W_hh       = (const float*)d_in[5];   // [3H,H]
  const float* b_ih       = (const float*)d_in[6];
  const float* b_hh       = (const float*)d_in[7];
  const float* W_out      = (const float*)d_in[8];   // [V,H]
  const float* b_out      = (const float*)d_in[9];
  float* out = (float*)d_out;

  char* ws = (char*)d_ws;
  size_t off = 0;
  auto take = [&](size_t bytes) -> char* {
    char* p = ws + off;
    off = (off + bytes + 255) & ~(size_t)255;
    return p;
  };
  bf16*  Wih_bf  = (bf16*) take((size_t)3 * H_ * E_ * sizeof(bf16));
  bf16*  Whh_bf  = (bf16*) take((size_t)3 * H_ * H_ * sizeof(bf16));
  bf16*  Wout_bf = (bf16*) take((size_t)V_ * H_ * sizeof(bf16));
  bf16*  Xbf     = (bf16*) take((size_t)T_ * B_ * E_ * sizeof(bf16));
  float* Gi      = (float*)take((size_t)T_ * B_ * 3 * H_ * sizeof(float));
  float* hist_f  = (float*)take((size_t)(T_ + 1) * B_ * H_ * sizeof(float));
  bf16*  hist_b  = (bf16*) take((size_t)(T_ + 1) * B_ * H_ * sizeof(bf16));
  (void)ws_size; (void)in_sizes; (void)n_in; (void)out_size;

  // 1) convert weights to bf16
  k_f32_to_bf16<<<1024, 256, 0, stream>>>(W_ih,  Wih_bf,  (size_t)3 * H_ * E_);
  k_f32_to_bf16<<<1024, 256, 0, stream>>>(W_hh,  Whh_bf,  (size_t)3 * H_ * H_);
  k_f32_to_bf16<<<4096, 256, 0, stream>>>(W_out, Wout_bf, (size_t)V_ * H_);

  // 2) embed + relu all teacher-forced tokens
  k_embed<<<T_ * B_, 128, 0, stream>>>(target, embedding, Xbf);

  // 3) h0
  k_init_h<<<64, 512, 0, stream>>>(enc_hidden, hist_f, hist_b);

  // 4) Gi = X @ W_ih^T + b_ih for all steps (WMMA GEMM 4096x3072x512)
  k_gi_gemm<<<dim3(48 / 4, 4096 / 32), 128, 0, stream>>>(Xbf, Wih_bf, b_ih, Gi);

  // 5) serial GRU recurrence
  const size_t BH = (size_t)B_ * H_;
  for (int t = 0; t < T_; ++t) {
    k_gru_step<<<8, 256, 0, stream>>>(
        hist_b + (size_t)t * BH, hist_f + (size_t)t * BH,
        Whh_bf, b_hh, Gi + (size_t)t * B_ * 3 * H_,
        hist_f + (size_t)(t + 1) * BH, hist_b + (size_t)(t + 1) * BH);
  }

  // 6) output GEMM (4096 x 32000 x 1024) into d_out, then in-place log-softmax
  k_out_gemm<<<dim3(500 / 4, 4096 / 32), 128, 0, stream>>>(
      hist_b + BH, Wout_bf, b_out, out);
  k_logsoftmax<<<B_ * T_, 256, 0, stream>>>(out);

  // 7) decoder_hidden = h_T
  k_copy_f32<<<64, 512, 0, stream>>>(hist_f + (size_t)T_ * BH,
                                     out + (size_t)B_ * T_ * V_, BH);
}
